// GINBlock_6476810682403
// MI455X (gfx1250) — matrix operand order repaired and checked by reading, not compiled, not run on hardware
//
#include <hip/hip_runtime.h>

// ---------------------------------------------------------------------------
// GIN block for MI455X (gfx1250, wave32):
//   agg = segment_sum(x[src], dst)            -> L2-resident f32 atomics
//   h   = relu((x+agg)@W1+b1)@W2+b2           -> v_wmma_f32_16x16x32_f16
//   out = LayerNorm(h)*gamma+beta + x@Wres+bres
// Weights are pre-swizzled once into a f16 WMMA-B-fragment image in scratch,
// so the hot GEMM loop does only coalesced b128 loads + wmma.
// ---------------------------------------------------------------------------

typedef _Float16 v16h __attribute__((ext_vector_type(16)));
typedef _Float16 v8h  __attribute__((ext_vector_type(8)));
typedef float    v8f  __attribute__((ext_vector_type(8)));

#define DIM       128
#define WAVES     4
#define H1STRIDE  136   // f16 row stride (128 + 8 pad) -> LDS conflict-free-ish
#define NFRAG     (3 * 4 * 8 * 32)   // matrices * ktiles * ntiles * lanes

// ---------------- kernel 1: zero the aggregation scratch -------------------
__global__ void zero_agg_kernel(float* __restrict__ p, int n4) {
    float4 z = make_float4(0.f, 0.f, 0.f, 0.f);
    for (int i = blockIdx.x * blockDim.x + threadIdx.x; i < n4;
         i += gridDim.x * blockDim.x)
        ((float4*)p)[i] = z;
}

// ---------------- kernel 2: edge scatter-add (one wave32 per edge) ---------
__global__ void scatter_edges_kernel(const float* __restrict__ x,
                                     const int* __restrict__ ei,
                                     float* __restrict__ agg, int E) {
    int e    = (blockIdx.x * blockDim.x + threadIdx.x) >> 5;  // wave id = edge
    int lane = threadIdx.x & 31;
    if (e >= E) return;
    int src = ei[e];          // wave-uniform -> scalar load
    int dst = ei[E + e];
    // 32 lanes x float4 = full 128-float row; x rows are L2-resident
    const float4 v = ((const float4*)(x + (size_t)src * DIM))[lane];
    float* d = agg + (size_t)dst * DIM + lane * 4;
    unsafeAtomicAdd(d + 0, v.x);   // global_atomic_add_f32, no return
    unsafeAtomicAdd(d + 1, v.y);
    unsafeAtomicAdd(d + 2, v.z);
    unsafeAtomicAdd(d + 3, v.w);
}

// ---------------- kernel 3: pre-swizzle weights into B-fragment image ------
// Fragment layout (ISA B 32x16, f16): lane holds column (lane&15) of ntile,
// half h -> K = kb+8h..kb+8h+7, then kb+16+8h..kb+16+8h+7.
// Image index: [m][kt][nt][lane] -> v16h (32 B), fully coalesced on read.
__global__ void wprep_kernel(const float* __restrict__ W1,
                             const float* __restrict__ W2,
                             const float* __restrict__ Wres,
                             v16h* __restrict__ wfrag) {
    int t = blockIdx.x * blockDim.x + threadIdx.x;
    if (t >= NFRAG) return;
    int lane = t & 31;
    int nt   = (t >> 5) & 7;
    int kt   = (t >> 8) & 3;
    int m    = t >> 10;
    const float* W = (m == 0) ? W1 : (m == 1) ? W2 : Wres;
    int col = nt * 16 + (lane & 15);
    int kb  = kt * 32 + 8 * (lane >> 4);
    v16h v;
#pragma unroll
    for (int i = 0; i < 16; ++i)
        v[i] = (_Float16)W[(kb + (i < 8 ? i : i + 8)) * DIM + col];
    wfrag[t] = v;
}

// ---------------- WMMA helpers ---------------------------------------------
__device__ __forceinline__ v16h load_b_frag(const v16h* __restrict__ wf,
                                            int m, int kt, int nt, int lane) {
    return wf[((m * 4 + kt) * 8 + nt) * 32 + lane];   // 2x global_load_b128
}

__device__ __forceinline__ v8f wmma_acc(v16h a, v16h b, v8f c) {
    return __builtin_amdgcn_wmma_f32_16x16x32_f16(
        /*neg_a=*/false, a, /*neg_b=*/false, b,
        /*c_mod=*/(short)0, c, /*reuse_a=*/false, /*reuse_b=*/false);
}

// ---------------- kernel 4: fused MLP + LayerNorm + residual ---------------
__global__ __launch_bounds__(WAVES * 32)
void gin_mlp_kernel(const float* __restrict__ x,   const float* __restrict__ agg,
                    const v16h* __restrict__ wf,
                    const float* __restrict__ b1,  const float* __restrict__ b2,
                    const float* __restrict__ gamma, const float* __restrict__ beta,
                    const float* __restrict__ bres,
                    float* __restrict__ out, int N) {
    __shared__ __align__(16) _Float16 lds[WAVES][16 * H1STRIDE];

    const int lane  = threadIdx.x & 31;
    const int wv    = threadIdx.x >> 5;
    const int hh    = lane >> 4;        // lane half (0/1)
    const int lcol  = lane & 15;
    const int tile0 = (blockIdx.x * WAVES + wv) * 16;

    // A-layout: this lane owns matrix row `lcol` of the tile.
    int row  = tile0 + lcol;
    int rowc = row < N ? row : N - 1;   // clamp: keep EXEC all-ones for WMMA
    const float* xr = x   + (size_t)rowc * DIM;
    const float* gr = agg + (size_t)rowc * DIM;

    // --- A fragments: a_x = f16(x), a_h = f16(x + agg)  (ISA 16-bit A 16x32)
    v16h a_x[4], a_h[4];
#pragma unroll
    for (int kt = 0; kt < 4; ++kt) {
        int kb = kt * 32 + 8 * hh;
        float4 x0 = *(const float4*)&xr[kb];
        float4 x1 = *(const float4*)&xr[kb + 4];
        float4 x2 = *(const float4*)&xr[kb + 16];
        float4 x3 = *(const float4*)&xr[kb + 20];
        float4 g0 = *(const float4*)&gr[kb];
        float4 g1 = *(const float4*)&gr[kb + 4];
        float4 g2 = *(const float4*)&gr[kb + 16];
        float4 g3 = *(const float4*)&gr[kb + 20];
        a_x[kt][0]  = (_Float16)x0.x; a_x[kt][1]  = (_Float16)x0.y;
        a_x[kt][2]  = (_Float16)x0.z; a_x[kt][3]  = (_Float16)x0.w;
        a_x[kt][4]  = (_Float16)x1.x; a_x[kt][5]  = (_Float16)x1.y;
        a_x[kt][6]  = (_Float16)x1.z; a_x[kt][7]  = (_Float16)x1.w;
        a_x[kt][8]  = (_Float16)x2.x; a_x[kt][9]  = (_Float16)x2.y;
        a_x[kt][10] = (_Float16)x2.z; a_x[kt][11] = (_Float16)x2.w;
        a_x[kt][12] = (_Float16)x3.x; a_x[kt][13] = (_Float16)x3.y;
        a_x[kt][14] = (_Float16)x3.z; a_x[kt][15] = (_Float16)x3.w;
        a_h[kt][0]  = (_Float16)(x0.x + g0.x); a_h[kt][1]  = (_Float16)(x0.y + g0.y);
        a_h[kt][2]  = (_Float16)(x0.z + g0.z); a_h[kt][3]  = (_Float16)(x0.w + g0.w);
        a_h[kt][4]  = (_Float16)(x1.x + g1.x); a_h[kt][5]  = (_Float16)(x1.y + g1.y);
        a_h[kt][6]  = (_Float16)(x1.z + g1.z); a_h[kt][7]  = (_Float16)(x1.w + g1.w);
        a_h[kt][8]  = (_Float16)(x2.x + g2.x); a_h[kt][9]  = (_Float16)(x2.y + g2.y);
        a_h[kt][10] = (_Float16)(x2.z + g2.z); a_h[kt][11] = (_Float16)(x2.w + g2.w);
        a_h[kt][12] = (_Float16)(x3.x + g3.x); a_h[kt][13] = (_Float16)(x3.y + g3.y);
        a_h[kt][14] = (_Float16)(x3.z + g3.z); a_h[kt][15] = (_Float16)(x3.w + g3.w);
    }

    // --- residual path: res = x @ Wres + bres  (kept register-resident)
    v8f res[8];
#pragma unroll
    for (int nt = 0; nt < 8; ++nt) {
        float bb = bres[nt * 16 + lcol];
        v8f c;
#pragma unroll
        for (int r = 0; r < 8; ++r) c[r] = bb;
#pragma unroll
        for (int kt = 0; kt < 4; ++kt)
            c = wmma_acc(a_x[kt], load_b_frag(wf, 2, kt, nt, lane), c);
        res[nt] = c;
    }

    // --- layer 1: h1 = relu((x+agg) @ W1 + b1) -> LDS f16 (row-major padded)
    _Float16* buf = lds[wv];
#pragma unroll
    for (int nt = 0; nt < 8; ++nt) {
        int col = nt * 16 + lcol;
        float bb = b1[col];
        v8f c;
#pragma unroll
        for (int r = 0; r < 8; ++r) c[r] = bb;
#pragma unroll
        for (int kt = 0; kt < 4; ++kt)
            c = wmma_acc(a_h[kt], load_b_frag(wf, 0, kt, nt, lane), c);
#pragma unroll
        for (int r = 0; r < 8; ++r) {
            float v = c[r] > 0.f ? c[r] : 0.f;                 // ReLU
            buf[(r + 8 * hh) * H1STRIDE + col] = (_Float16)v;  // m = r + 8*half
        }
    }
    __syncthreads();

    // --- reload h1 in A-fragment layout (transpose via LDS), already f16
    v16h a2[4];
#pragma unroll
    for (int kt = 0; kt < 4; ++kt) {
        int kb = kt * 32 + 8 * hh;
        const _Float16* rb = &buf[lcol * H1STRIDE];
        v8h lo = *(const v8h*)&rb[kb];        // 16 B aligned ds load
        v8h hi = *(const v8h*)&rb[kb + 16];
        a2[kt] = __builtin_shufflevector(lo, hi, 0, 1, 2, 3, 4, 5, 6, 7,
                                                 8, 9, 10, 11, 12, 13, 14, 15);
    }

    // --- layer 2: h2 = h1 @ W2 + b2   (register-resident, D-layout)
    v8f h2[8];
#pragma unroll
    for (int nt = 0; nt < 8; ++nt) {
        float bb = b2[nt * 16 + lcol];
        v8f c;
#pragma unroll
        for (int r = 0; r < 8; ++r) c[r] = bb;
#pragma unroll
        for (int kt = 0; kt < 4; ++kt)
            c = wmma_acc(a2[kt], load_b_frag(wf, 1, kt, nt, lane), c);
        h2[nt] = c;
    }

    // --- LayerNorm across 128 cols; rows live in D-layout.
    // Row (r + 8*half) sits in the 16 lanes of this lane-half, one col per nt.
    // xor-shuffles 1,2,4,8 reduce strictly within the half.
    v8f mu, rs;
#pragma unroll
    for (int r = 0; r < 8; ++r) {
        float s = 0.f;
#pragma unroll
        for (int nt = 0; nt < 8; ++nt) s += h2[nt][r];
        s += __shfl_xor(s, 1); s += __shfl_xor(s, 2);
        s += __shfl_xor(s, 4); s += __shfl_xor(s, 8);
        float m = s * (1.0f / 128.0f);
        float q = 0.f;
#pragma unroll
        for (int nt = 0; nt < 8; ++nt) {
            float d = h2[nt][r] - m;
            q += d * d;
        }
        q += __shfl_xor(q, 1); q += __shfl_xor(q, 2);
        q += __shfl_xor(q, 4); q += __shfl_xor(q, 8);
        mu[r] = m;
        rs[r] = rsqrtf(q * (1.0f / 128.0f) + 1e-5f);
    }

    // --- out = LN(h2)*gamma + beta + res   (predicated scalar stores)
#pragma unroll
    for (int nt = 0; nt < 8; ++nt) {
        int col = nt * 16 + lcol;
        float g = gamma[col], bt = beta[col];
#pragma unroll
        for (int r = 0; r < 8; ++r) {
            int m = tile0 + r + 8 * hh;
            if (m < N)
                out[(size_t)m * DIM + col] =
                    (h2[nt][r] - mu[r]) * rs[r] * g + bt + res[nt][r];
        }
    }
}

// ---------------------------------------------------------------------------
extern "C" void kernel_launch(void* const* d_in, const int* in_sizes, int n_in,
                              void* d_out, int out_size, void* d_ws, size_t ws_size,
                              hipStream_t stream) {
    const float* x     = (const float*)d_in[0];
    const int*   ei    = (const int*)d_in[1];   // edge_index, [2, E]
    const float* W1    = (const float*)d_in[2];
    const float* b1    = (const float*)d_in[3];
    const float* W2    = (const float*)d_in[4];
    const float* b2    = (const float*)d_in[5];
    const float* gamma = (const float*)d_in[6];
    const float* beta  = (const float*)d_in[7];
    const float* Wres  = (const float*)d_in[8];
    const float* bres  = (const float*)d_in[9];
    float* out = (float*)d_out;

    const int N = in_sizes[0] / DIM;
    const int E = in_sizes[1] / 2;

    // scratch layout: [ agg: N*128 f32 ][ wfrag: 3*128*128 f16 fragment image ]
    float* agg = (float*)d_ws;
    v16h* wfrag = (v16h*)((char*)d_ws + (size_t)N * DIM * sizeof(float));

    // 1) zero agg
    zero_agg_kernel<<<2048, 256, 0, stream>>>(agg, N * (DIM / 4));

    // 2) pre-swizzle weights into f16 B-fragment image (one-time, tiny)
    wprep_kernel<<<(NFRAG + 255) / 256, 256, 0, stream>>>(W1, W2, Wres, wfrag);

    // 3) scatter-add: one wave32 per edge, 8 edges per 256-thread block
    int sblocks = (E + 7) / 8;
    scatter_edges_kernel<<<sblocks, 256, 0, stream>>>(x, ei, agg, E);

    // 4) fused MLP + LN + residual: 16 nodes per wave, 64 per block
    int gblocks = (N + WAVES * 16 - 1) / (WAVES * 16);
    gin_mlp_kernel<<<gblocks, WAVES * 32, 0, stream>>>(
        x, agg, wfrag, b1, b2, gamma, beta, bres, out, N);
}
